// MoEBlock_51883204935735
// MI455X (gfx1250) — matrix operand order repaired and checked by reading, not compiled, not run on hardware
//
#include <hip/hip_runtime.h>
#include <hip/hip_bf16.h>
#include <math.h>

// ---------------------------------------------------------------------------
// MoE block for MI455X (gfx1250, wave32, WMMA, TDM tensor_load_to_lds).
// Sizes fixed by the reference: D=1024, H=4096, E=8, K=2, B=4, T=1024.
// ---------------------------------------------------------------------------

#define D_DIM 1024
#define H_DIM 4096
#define E_NUM 8
#define N_TOK 4096          // B*T
#define LN_EPS 1e-5f

typedef __attribute__((ext_vector_type(16))) __bf16 v16bf;
typedef __attribute__((ext_vector_type(8)))  __bf16 v8bf;
typedef __attribute__((ext_vector_type(4)))  __bf16 v4bf;
typedef __attribute__((ext_vector_type(8)))  float  v8f;
typedef __attribute__((ext_vector_type(4)))  unsigned int v4u;
typedef __attribute__((ext_vector_type(8)))  int v8i;
typedef __attribute__((ext_vector_type(4)))  int v4i;

// ---------------------------------------------------------------------------
// prep: out = x (residual accumulator init), xb = bf16(x)
// ---------------------------------------------------------------------------
__global__ __launch_bounds__(256)
void prep_kernel(const float* __restrict__ x, __bf16* __restrict__ xb,
                 float* __restrict__ out) {
  size_t i = ((size_t)blockIdx.x * blockDim.x + threadIdx.x) * 4;
  float4 v = *(const float4*)(x + i);
  *(float4*)(out + i) = v;
  v4bf b;
  b[0] = (__bf16)v.x; b[1] = (__bf16)v.y;
  b[2] = (__bf16)v.z; b[3] = (__bf16)v.w;
  *(v4bf*)(xb + i) = b;
}

// ---------------------------------------------------------------------------
// router: logits = x@Wr + br + noise ; softmax ; top-2 ; comb[N,E]
// ---------------------------------------------------------------------------
__global__ __launch_bounds__(128)
void router_kernel(const float* __restrict__ x, const float* __restrict__ noise,
                   const float* __restrict__ Wr, const float* __restrict__ br,
                   float* __restrict__ comb) {
  const int n = blockIdx.x;
  const int t = threadIdx.x;
  __shared__ float red[E_NUM][128];

  float p[E_NUM];
#pragma unroll
  for (int e = 0; e < E_NUM; ++e) p[e] = 0.0f;

  const float* xr = x + (size_t)n * D_DIM;
  for (int i = t; i < D_DIM; i += 128) {
    float xv = xr[i];
    const float* wr = Wr + (size_t)i * E_NUM;
#pragma unroll
    for (int e = 0; e < E_NUM; ++e) p[e] += xv * wr[e];
  }
#pragma unroll
  for (int e = 0; e < E_NUM; ++e) red[e][t] = p[e];
  __syncthreads();
  for (int s = 64; s > 0; s >>= 1) {
    if (t < s) {
#pragma unroll
      for (int e = 0; e < E_NUM; ++e) red[e][t] += red[e][t + s];
    }
    __syncthreads();
  }
  if (t == 0) {
    float lg[E_NUM];
    float mx = -3.4e38f;
#pragma unroll
    for (int e = 0; e < E_NUM; ++e) {
      lg[e] = red[e][0] + br[e] + noise[(size_t)n * E_NUM + e];
      mx = fmaxf(mx, lg[e]);
    }
    float s = 0.0f;
#pragma unroll
    for (int e = 0; e < E_NUM; ++e) { lg[e] = expf(lg[e] - mx); s += lg[e]; }
    float inv = 1.0f / s;
#pragma unroll
    for (int e = 0; e < E_NUM; ++e) lg[e] *= inv;
    int i1 = 0;
#pragma unroll
    for (int e = 1; e < E_NUM; ++e) if (lg[e] > lg[i1]) i1 = e;
    int i2 = (i1 == 0) ? 1 : 0;
#pragma unroll
    for (int e = 0; e < E_NUM; ++e)
      if (e != i1 && lg[e] > lg[i2]) i2 = e;
#pragma unroll
    for (int e = 0; e < E_NUM; ++e)
      comb[(size_t)n * E_NUM + e] = (e == i1 || e == i2) ? lg[e] : 0.0f;
  }
}

// ---------------------------------------------------------------------------
// Fragment assembly from LDS (two 16B ds_load_b128-sized reads).
// A-matrix 16x32 bf16 layout (ISA 7.12.2): lane L holds row M=L%16; element i of
// the v16bf maps to K = (i<8 ? hs*8+i : 16+hs*8+(i-8)), hs = L/16.
// ---------------------------------------------------------------------------
__device__ inline v16bf load_frag(const __bf16* p0, const __bf16* p1) {
  union { v16bf v; v8bf h[2]; } u;
  u.h[0] = *(const v8bf*)p0;
  u.h[1] = *(const v8bf*)p1;
  return u.v;
}

__device__ inline unsigned pack_bf16(float a, float b) {
  union { __bf16 h[2]; unsigned u; } q;
  q.h[0] = (__bf16)a;
  q.h[1] = (__bf16)b;
  return q.u;
}

// ---------------------------------------------------------------------------
// Tiled bf16 WMMA GEMM with fused epilogue, double-buffered LDS.
// A tile (already bf16) is moved global->LDS by the Tensor Data Mover:
// one tensor_load_to_lds per tile per workgroup (wave 0), with the D#'s
// pad fields producing the 72-element padded LDS row stride directly.
// B tile (fp32) is batch-loaded, converted, and stored transposed+packed.
//   C[M,Nn] = A[M,Kdim](bf16) @ Bw[Kdim,Nn](fp32 -> bf16 on stage)
// MODE 0: h = GELU_exact(C + bias)           -> outH   (bf16, for GEMM2)
// MODE 1: out[n,d] += comb[n,e]*(C + bias)   -> outAcc (fp32 residual accum)
// Block tile 128x128, BK=64; 8 waves, each owns a 64x32 sub-tile
// (4x2 grid of v_wmma_f32_16x16x32_bf16, 64 accumulator VGPRs/lane).
// ---------------------------------------------------------------------------
template <int MODE>
__global__ __launch_bounds__(256)
void moe_gemm_kernel(const __bf16* __restrict__ A, const float* __restrict__ Bw,
                     const float* __restrict__ bias, int Kdim, int Nn,
                     __bf16* __restrict__ outH, float* __restrict__ outAcc,
                     const float* __restrict__ comb, int e) {
  __shared__ __align__(16) __bf16 As[2][128][72];  // [buf][token][K] (+pad)
  __shared__ __align__(16) __bf16 Bs[2][128][72];  // [buf][col N][K] (B^T)

  const int t     = threadIdx.x;
  const int wid   = t >> 5;
  const int lane  = t & 31;
  const int waveM = wid >> 2;      // 0..1  -> 64 rows each
  const int waveN = wid & 3;       // 0..3  -> 32 cols each
  const int hs    = lane >> 4;     // lane half-select
  const int l16   = lane & 15;

  const int rowBase = blockIdx.y * 128;
  const int colBase = blockIdx.x * 128;

  v8f acc[4][2] = {};

  // --- TDM: DMA the 128x64 bf16 A tile into As[b] with padded rows. ---
  // D# group1 fields: data_size=2B (code 1); pad_enable; pad_interval code 4
  // (= 32 dwords = one 128B tile row); pad_amount code 3 (= 4 dwords = 16B,
  // i.e. 72-element LDS row stride); tile 64x128; dim0 stride = Kdim.
  auto stage_a_tdm = [&](int kt, int b) {
    unsigned lds = (unsigned)(size_t)&As[b][0][0];
    unsigned long long g =
        (unsigned long long)(A + (size_t)rowBase * Kdim + (size_t)kt * 64);
    v4u g0;
    g0[0] = 1u;                                   // count=1, user descriptor
    g0[1] = lds;                                  // lds_addr
    g0[2] = (unsigned)g;                          // global_addr[31:0]
    g0[3] = (unsigned)(g >> 32) | (2u << 30);     // global_addr[56:32], type=2
    v8i g1;
    g1[0] = (1 << 16) | (1 << 20) | (4 << 22) | (3 << 25);
    g1[1] = (int)((Kdim & 0xFFFF) << 16);         // tensor_dim0 lo16
    g1[2] = (int)(((unsigned)Kdim >> 16) | ((N_TOK & 0xFFFF) << 16));
    g1[3] = (int)((N_TOK >> 16) | (64 << 16));    // tensor_dim1 hi, tile_dim0
    g1[4] = 128;                                  // tile_dim1=128, tile_dim2=0
    g1[5] = Kdim;                                 // tensor_dim0_stride lo32
    g1[6] = 0;
    g1[7] = 0;
    v4i z4 = {0, 0, 0, 0};                        // groups 2/3: 2-D tile
    v8i z8 = {0, 0, 0, 0, 0, 0, 0, 0};
    __builtin_amdgcn_tensor_load_to_lds(g0, g1, z4, z4, z8, 0);
  };

  // --- B tile: 64(K)x128(N) fp32; batch loads, then cvt+packed transpose ---
  auto stage_b = [&](int kt, int b) {
    float4 f0[4], f1[4];
#pragma unroll
    for (int i = 0; i < 4; ++i) {
      int idx = t + i * 256;
      int k = (idx >> 5) * 2;          // pair of K rows
      int c = (idx & 31) * 4;          // 4 N columns
      const float* s0 = Bw + (size_t)(kt * 64 + k) * Nn + colBase + c;
      f0[i] = *(const float4*)s0;
      f1[i] = *(const float4*)(s0 + Nn);
    }
#pragma unroll
    for (int i = 0; i < 4; ++i) {
      int idx = t + i * 256;
      int k = (idx >> 5) * 2;
      int c = (idx & 31) * 4;
      const float* a0 = &f0[i].x;
      const float* a1 = &f1[i].x;
#pragma unroll
      for (int j = 0; j < 4; ++j) {
        unsigned u = pack_bf16(a0[j], a1[j]);   // (k, k+1) packed
        *(unsigned*)&Bs[b][c + j][k] = u;
      }
    }
  };

  const int kIters = Kdim >> 6;
  if (wid == 0) stage_a_tdm(0, 0);
  stage_b(0, 0);
  int buf = 0;
  for (int kt = 0; kt < kIters; ++kt) {
    if (wid == 0) __builtin_amdgcn_s_wait_tensorcnt(0);  // A tile landed
    __syncthreads();                                     // publish A + B tiles
    if (kt + 1 < kIters) {                               // prefetch next tile
      if (wid == 0) stage_a_tdm(kt + 1, buf ^ 1);
      stage_b(kt + 1, buf ^ 1);
    }

    // ---- two K=32 WMMA steps on current buffer ----
#pragma unroll
    for (int ks = 0; ks < 64; ks += 32) {
      v16bf af[4], bfr[2];
#pragma unroll
      for (int m = 0; m < 4; ++m) {
        int row = waveM * 64 + m * 16 + l16;
        af[m] = load_frag(&As[buf][row][ks + hs * 8],
                          &As[buf][row][ks + 16 + hs * 8]);
      }
#pragma unroll
      for (int nn = 0; nn < 2; ++nn) {
        int col = waveN * 32 + nn * 16 + l16;
        bfr[nn] = load_frag(&Bs[buf][col][ks + hs * 8],
                            &Bs[buf][col][ks + 16 + hs * 8]);
      }
#pragma unroll
      for (int m = 0; m < 4; ++m)
#pragma unroll
        for (int nn = 0; nn < 2; ++nn)
          acc[m][nn] = __builtin_amdgcn_wmma_f32_16x16x32_bf16(
              false, af[m], false, bfr[nn], (short)0, acc[m][nn], false, false);
    }
    buf ^= 1;
  }

  // ---- epilogue: C layout: lane L -> col=L%16, VGPR r -> row r+8*(L/16) ----
#pragma unroll
  for (int m = 0; m < 4; ++m) {
#pragma unroll
    for (int nn = 0; nn < 2; ++nn) {
      int col = colBase + waveN * 32 + nn * 16 + l16;
      float bv = bias[col];
#pragma unroll
      for (int r = 0; r < 8; ++r) {
        int row = rowBase + waveM * 64 + m * 16 + hs * 8 + r;
        float v = acc[m][nn][r] + bv;
        if (MODE == 0) {
          // exact (erf) GELU, matching torch/jax approximate=False
          v = 0.5f * v * (1.0f + erff(v * 0.70710678118654752f));
          outH[(size_t)row * Nn + col] = (__bf16)v;
        } else {
          float w = comb[(size_t)row * E_NUM + e];
          float* p = outAcc + (size_t)row * Nn + col;
          *p += w * v;  // unique (row,col) per launch; stream-serialized
        }
      }
    }
  }
}

// ---------------------------------------------------------------------------
// LayerNorm over D=1024, in place on out (which already holds x + moe_out).
// ---------------------------------------------------------------------------
__global__ __launch_bounds__(256)
void ln_kernel(float* __restrict__ out, const float* __restrict__ gamma,
               const float* __restrict__ beta) {
  const int n = blockIdx.x;
  const int t = threadIdx.x;
  __shared__ float s1[256];
  __shared__ float s2[256];
  float* row = out + (size_t)n * D_DIM;

  float vals[4];
  float a = 0.0f, b = 0.0f;
#pragma unroll
  for (int i = 0; i < 4; ++i) {
    float v = row[t + i * 256];
    vals[i] = v;
    a += v;
    b += v * v;
  }
  s1[t] = a; s2[t] = b;
  __syncthreads();
  for (int s = 128; s > 0; s >>= 1) {
    if (t < s) { s1[t] += s1[t + s]; s2[t] += s2[t + s]; }
    __syncthreads();
  }
  float mu  = s1[0] * (1.0f / D_DIM);
  float var = s2[0] * (1.0f / D_DIM) - mu * mu;
  float inv = rsqrtf(var + LN_EPS);
#pragma unroll
  for (int i = 0; i < 4; ++i) {
    int c = t + i * 256;
    row[c] = (vals[i] - mu) * inv * gamma[c] + beta[c];
  }
}

// ---------------------------------------------------------------------------
// Launch: prep -> router -> per-expert (GEMM1+GELU, GEMM2+combine) -> LN
// Workspace: comb (128KB) | xb bf16 (8MB) | hb bf16 (32MB)
// ---------------------------------------------------------------------------
extern "C" void kernel_launch(void* const* d_in, const int* in_sizes, int n_in,
                              void* d_out, int out_size, void* d_ws,
                              size_t ws_size, hipStream_t stream) {
  const float* x     = (const float*)d_in[0];
  const float* noise = (const float*)d_in[1];
  const float* Wr    = (const float*)d_in[2];
  const float* br    = (const float*)d_in[3];
  const float* W1    = (const float*)d_in[4];
  const float* b1    = (const float*)d_in[5];
  const float* W2    = (const float*)d_in[6];
  const float* b2    = (const float*)d_in[7];
  const float* gamma = (const float*)d_in[8];
  const float* beta  = (const float*)d_in[9];
  float* out = (float*)d_out;

  char* ws = (char*)d_ws;
  float*  comb = (float*)ws;                              // 4096*8*4
  __bf16* xb   = (__bf16*)(ws + 131072);                  // 4096*1024*2
  __bf16* hb   = (__bf16*)(ws + 131072 + 8388608);        // 4096*4096*2

  prep_kernel<<<(N_TOK * D_DIM) / 1024, 256, 0, stream>>>(x, xb, out);
  router_kernel<<<N_TOK, 128, 0, stream>>>(x, noise, Wr, br, comb);

  for (int e = 0; e < E_NUM; ++e) {
    moe_gemm_kernel<0><<<dim3(H_DIM / 128, N_TOK / 128), 256, 0, stream>>>(
        xb, W1 + (size_t)e * D_DIM * H_DIM, b1 + (size_t)e * H_DIM,
        D_DIM, H_DIM, hb, nullptr, nullptr, e);
    moe_gemm_kernel<1><<<dim3(D_DIM / 128, N_TOK / 128), 256, 0, stream>>>(
        hb, W2 + (size_t)e * H_DIM * D_DIM, b2 + (size_t)e * D_DIM,
        H_DIM, D_DIM, nullptr, out, comb, e);
  }

  ln_kernel<<<N_TOK, 256, 0, stream>>>(out, gamma, beta);
}
